// PPFNet_75204877353245
// MI455X (gfx1250) — matrix operand order repaired and checked by reading, not compiled, not run on hardware
//
#include <hip/hip_runtime.h>
#include <hip/hip_bf16.h>
#include <math.h>

#define Bc   8
#define N0   4096
#define KNNK 16
#define Hd   128
#define OUTC 40

typedef _Float16 f16;
typedef __attribute__((ext_vector_type(16))) _Float16 v16h;
typedef __attribute__((ext_vector_type(8)))  _Float16 v8h;
typedef __attribute__((ext_vector_type(8)))  float    v8f;
typedef __attribute__((__vector_size__(16))) int      v4i;

#define AS1 __attribute__((address_space(1)))
#define AS3 __attribute__((address_space(3)))

// ---------------------------------------------------------------------------
// A-fragment (f16 16x32, row-major source, strideF16 elements per row).
// Lane l: row = l&15, half = l>>4.  VGPR dwords 0..3 hold K = kbase+8*half+{0..7},
// dwords 4..7 hold K = kbase+16+8*half+{0..7}  -> two 16B loads.
__device__ inline v16h load_A(const f16* base, int strideF16, int kbase, int lane) {
    const int half = lane >> 4, row = lane & 15;
    const f16* r = base + row * strideF16 + kbase + 8 * half;
    v8h lo = *(const v8h*)r;
    v8h hi = *(const v8h*)(r + 16);
    return __builtin_shufflevector(lo, hi, 0,1,2,3,4,5,6,7,8,9,10,11,12,13,14,15);
}

// B-fragment (f16 32x16) from transposed weights wT[N][KP] (row-major).
// Lane l: N = 16*t + (l&15), K = kbase + 8*(l>>4) + {0..7} and kbase + 16 + 8*(l>>4) + {0..7} -> two 16B loads.
__device__ inline v16h load_B(const f16* wT, int KP, int t, int kbase, int lane) {
    const f16* r = wT + (size_t)(16 * t + (lane & 15)) * KP + kbase + 8 * (lane >> 4);
    v8h lo = *(const v8h*)r;
    v8h hi = *(const v8h*)(r + 16);
    return __builtin_shufflevector(lo, hi, 0,1,2,3,4,5,6,7,8,9,10,11,12,13,14,15);
}

// ---------------------------------------------------------------------------
// Convert f32 weight [Kin][128] -> f16 transposed [128][KP], zero-padded K.
__global__ void prep_w_kernel(const float* __restrict__ w, f16* __restrict__ wT,
                              int Kin, int KP) {
    int t = blockIdx.x * 256 + threadIdx.x;
    if (t >= 128 * KP) return;
    int nIdx = t / KP, k = t - nIdx * KP;
    wT[t] = (k < Kin) ? (f16)w[k * 128 + nIdx] : (f16)0.f;
}

// ---------------------------------------------------------------------------
// Brute-force KNN within each cloud (self excluded). One thread per query,
// LDS-tiled candidates, register-resident sorted top-16 (compare-exchange).
__global__ __launch_bounds__(256) void knn_kernel(const float* __restrict__ p,
                                                  int n, int* __restrict__ out) {
    const int per = n / 256;
    const int b = blockIdx.x / per;
    const int i = (blockIdx.x % per) * 256 + threadIdx.x;
    const float* pc = p + (size_t)b * n * 3;
    const float px = pc[i*3+0], py = pc[i*3+1], pz = pc[i*3+2];

    float bd[KNNK]; int bi[KNNK];
#pragma unroll
    for (int k = 0; k < KNNK; ++k) { bd[k] = 3.4e38f; bi[k] = 0; }

    __shared__ float tile[256 * 3];
    for (int t0 = 0; t0 < n; t0 += 256) {
        int jl = t0 + threadIdx.x;
        tile[threadIdx.x*3+0] = pc[jl*3+0];
        tile[threadIdx.x*3+1] = pc[jl*3+1];
        tile[threadIdx.x*3+2] = pc[jl*3+2];
        __syncthreads();
        for (int jj = 0; jj < 256; ++jj) {
            int j = t0 + jj;
            float dx = tile[jj*3+0] - px, dy = tile[jj*3+1] - py, dz = tile[jj*3+2] - pz;
            float d2 = dx*dx + dy*dy + dz*dz;
            if (j != i && d2 < bd[KNNK-1]) {
                float cd = d2; int ci = j;
#pragma unroll
                for (int k = 0; k < KNNK; ++k) {        // sorted insert, no dyn idx
                    bool sw = cd < bd[k];
                    float td = sw ? bd[k] : cd;  int ti = sw ? bi[k] : ci;
                    bd[k] = sw ? cd : bd[k];     bi[k] = sw ? ci : bi[k];
                    cd = td; ci = ti;
                }
            }
        }
        __syncthreads();
    }
#pragma unroll
    for (int k = 0; k < KNNK; ++k) out[((size_t)b * n + i) * KNNK + k] = bi[k];
}

// ---------------------------------------------------------------------------
// Farthest point sampling, one block per cloud. Points staged in LDS; dmin in
// registers; argmax via wave shuffles + one cross-wave stage (2 barriers/step).
// Tie-break matches jnp.argmax (largest value, then smallest index).
__global__ __launch_bounds__(1024) void fps_kernel(const float* __restrict__ p,
                                                   int n, int m, int* __restrict__ sel) {
    const int b = blockIdx.x, tid = threadIdx.x;
    const int lane = tid & 31, wv = tid >> 5;
    const int P = n >> 10;                      // points per thread (4 or 2)
    __shared__ float pl[N0 * 3];
    __shared__ float wvv[32];
    __shared__ int   wvi[32];
    __shared__ int   s_last;
    for (int t = tid; t < n * 3; t += 1024) pl[t] = p[(size_t)b * n * 3 + t];
    if (tid == 0) { s_last = 0; sel[(size_t)b * m] = 0; }
    __syncthreads();

    float dmin[4], ptx[4], pty[4], ptz[4];
#pragma unroll
    for (int k = 0; k < 4; ++k) {
        if (k < P) {
            int i = tid + (k << 10);
            ptx[k] = pl[i*3+0]; pty[k] = pl[i*3+1]; ptz[k] = pl[i*3+2];
        }
        dmin[k] = 3.4e38f;
    }

    for (int s = 1; s < m; ++s) {
        const int last = s_last;
        const float lx = pl[last*3+0], ly = pl[last*3+1], lz = pl[last*3+2];
        float bv = -1.f; int bix = 0;
#pragma unroll
        for (int k = 0; k < 4; ++k) {
            if (k < P) {
                float dx = ptx[k]-lx, dy = pty[k]-ly, dz = ptz[k]-lz;
                float dm = fminf(dmin[k], dx*dx + dy*dy + dz*dz);
                dmin[k] = dm;
                if (dm > bv) { bv = dm; bix = tid + (k << 10); }  // k asc => idx asc
            }
        }
#pragma unroll
        for (int o = 16; o > 0; o >>= 1) {                 // intra-wave argmax
            float v2 = __shfl_xor(bv, o, 32);
            int   i2 = __shfl_xor(bix, o, 32);
            if (v2 > bv || (v2 == bv && i2 < bix)) { bv = v2; bix = i2; }
        }
        if (lane == 0) { wvv[wv] = bv; wvi[wv] = bix; }
        __syncthreads();
        if (wv == 0) {                                     // cross-wave argmax
            float v = wvv[lane]; int ii = wvi[lane];
#pragma unroll
            for (int o = 16; o > 0; o >>= 1) {
                float v2 = __shfl_xor(v, o, 32);
                int   i2 = __shfl_xor(ii, o, 32);
                if (v2 > v || (v2 == v && i2 < ii)) { v = v2; ii = i2; }
            }
            if (lane == 0) { s_last = ii; sel[(size_t)b * m + s] = ii; }
        }
        __syncthreads();
    }
}

// ---------------------------------------------------------------------------
__global__ __launch_bounds__(128) void gather_kernel(
    const float* __restrict__ p, const float* __restrict__ nrm,
    const float* __restrict__ x, const int* __restrict__ sel, int n, int m,
    float* __restrict__ po, float* __restrict__ no, float* __restrict__ xo) {
    const int node = blockIdx.x;          // b*m + i
    const int b = node / m;
    const int s = sel[node];
    const size_t src = (size_t)b * n + s;
    if (threadIdx.x < 3) {
        po[(size_t)node*3 + threadIdx.x] = p[src*3 + threadIdx.x];
        no[(size_t)node*3 + threadIdx.x] = nrm[src*3 + threadIdx.x];
    }
    xo[(size_t)node*Hd + threadIdx.x] = x[src*Hd + threadIdx.x];
}

// ---------------------------------------------------------------------------
// PPFConv: one wave per node; its 16 KNN edges = the WMMA M dimension.
// GEMM2 weights staged block-wide in LDS (async global->LDS when available);
// h1 overlays the feat buffer to stay within the static LDS budget.
__global__ __launch_bounds__(128) void ppf_conv_kernel(
    const float* __restrict__ p, const float* __restrict__ nrm,
    const float* __restrict__ xin, const int* __restrict__ knn,
    const f16* __restrict__ w1T, int nchunks, const float* __restrict__ b1,
    const f16* __restrict__ w2T, const float* __restrict__ b2,
    float* __restrict__ xout, int n) {
    __shared__ __align__(16) f16 fbuf[4][16][160];   // feat (stride 160) then h1 (stride 128)
    __shared__ __align__(16) f16 w2l[Hd * Hd];       // staged GEMM2 weights (32KB)
    const int wv = threadIdx.x >> 5;
    const int lane = threadIdx.x & 31;
    const int node = blockIdx.x * 4 + wv;
    const int b = node / n;
    const size_t gi = (size_t)node;
    const int KP1 = nchunks * 32;
    const int xbase = xin ? Hd : 0;

    // ---- stage w2T into LDS (whole block) -------------------------------
#if defined(__AMDGCN__) && __has_builtin(__builtin_amdgcn_global_load_async_to_lds_b128)
    for (int t = threadIdx.x; t < (Hd * Hd) / 8; t += 128) {
        v4i* gsrc = (v4i*)(const void*)(w2T + (size_t)t * 8);   // drop const, retype
        v4i* ldst = (v4i*)(void*)(&w2l[(size_t)t * 8]);
        __builtin_amdgcn_global_load_async_to_lds_b128(
            (AS1 v4i*)gsrc, (AS3 v4i*)ldst, 0, 0);
    }
    asm volatile("s_wait_asynccnt 0" ::: "memory");
#else
    {
        const v8h* src = (const v8h*)w2T;
        v8h* dst = (v8h*)w2l;
        for (int t = threadIdx.x; t < (Hd * Hd) / 8; t += 128) dst[t] = src[t];
    }
#endif

    // ---- build edge features in LDS (per wave, no block sync needed) ----
    unsigned int* fz = (unsigned int*)&fbuf[wv][0][0];
#pragma unroll
    for (int t = 0; t < (16 * 160 / 2) / 32; ++t) fz[lane + t * 32] = 0u;
    asm volatile("" ::: "memory");

    if (lane < 16) {                      // one edge per lane: PPF features
        const int e = lane;
        const int j = knn[gi * KNNK + e];
        const size_t gj = (size_t)b * n + j;
        float pix = p[gi*3+0], piy = p[gi*3+1], piz = p[gi*3+2];
        float nix = nrm[gi*3+0], niy = nrm[gi*3+1], niz = nrm[gi*3+2];
        float dx = p[gj*3+0]-pix, dy = p[gj*3+1]-piy, dz = p[gj*3+2]-piz;
        float njx = nrm[gj*3+0], njy = nrm[gj*3+1], njz = nrm[gj*3+2];
        float dn = sqrtf(dx*dx + dy*dy + dz*dz);
        float cx = niy*dz - niz*dy, cy = niz*dx - nix*dz, cz = nix*dy - niy*dx;
        float a1 = atan2f(sqrtf(cx*cx+cy*cy+cz*cz), nix*dx + niy*dy + niz*dz);
        cx = njy*dz - njz*dy; cy = njz*dx - njx*dz; cz = njx*dy - njy*dx;
        float a2 = atan2f(sqrtf(cx*cx+cy*cy+cz*cz), njx*dx + njy*dy + njz*dz);
        cx = niy*njz - niz*njy; cy = niz*njx - nix*njz; cz = nix*njy - niy*njx;
        float a3 = atan2f(sqrtf(cx*cx+cy*cy+cz*cz), nix*njx + niy*njy + niz*njz);
        fbuf[wv][e][xbase+0] = (f16)dn; fbuf[wv][e][xbase+1] = (f16)a1;
        fbuf[wv][e][xbase+2] = (f16)a2; fbuf[wv][e][xbase+3] = (f16)a3;
    }
    if (xin) {                            // gather neighbor features -> f16 LDS
        for (int e = 0; e < KNNK; ++e) {
            const int j = knn[gi * KNNK + e];
            const float* xr = xin + ((size_t)b * n + j) * Hd;
#pragma unroll
            for (int c = 0; c < Hd / 32; ++c)
                fbuf[wv][e][lane + c * 32] = (f16)xr[lane + c * 32];
        }
    }
    __syncthreads();                      // w2l staged + feat visible (wave lockstep)
    asm volatile("s_wait_dscnt 0" ::: "memory");

    v8f acc[8];
#pragma unroll
    for (int t = 0; t < 8; ++t)
#pragma unroll
        for (int q = 0; q < 8; ++q) acc[t][q] = 0.f;

    for (int c = 0; c < nchunks; ++c) {            // GEMM1: [16,KP1]x[KP1,128]
        v16h a = load_A(&fbuf[wv][0][0], 160, c * 32, lane);
#pragma unroll
        for (int t = 0; t < 8; ++t) {
            v16h bb = load_B(w1T, KP1, t, c * 32, lane);
            acc[t] = __builtin_amdgcn_wmma_f32_16x16x32_f16(
                false, a, false, bb, (short)0, acc[t], false, false);
        }
    }

    // ---- bias + relu -> h1 (overlays feat region, stride 128) ----------
    f16* h1w = &fbuf[wv][0][0];
    const int half = lane >> 4, col = lane & 15;
    asm volatile("" ::: "memory");
#pragma unroll
    for (int t = 0; t < 8; ++t) {
        const float bias = b1[t * 16 + col];
#pragma unroll
        for (int v = 0; v < 8; ++v)
            h1w[(v + 8 * half) * Hd + t * 16 + col] = (f16)fmaxf(acc[t][v] + bias, 0.f);
    }
    asm volatile("s_wait_dscnt 0" ::: "memory");

#pragma unroll
    for (int t = 0; t < 8; ++t)
#pragma unroll
        for (int q = 0; q < 8; ++q) acc[t][q] = 0.f;

#pragma unroll
    for (int c = 0; c < 4; ++c) {                  // GEMM2: [16,128]x[128,128]
        v16h a = load_A(h1w, Hd, c * 32, lane);
#pragma unroll
        for (int t = 0; t < 8; ++t) {
            v16h bb = load_B(w2l, Hd, t, c * 32, lane);   // from LDS
            acc[t] = __builtin_amdgcn_wmma_f32_16x16x32_f16(
                false, a, false, bb, (short)0, acc[t], false, false);
        }
    }

#pragma unroll
    for (int t = 0; t < 8; ++t) {                  // max over K, +b2, relu
        float m = acc[t][0];
#pragma unroll
        for (int v = 1; v < 8; ++v) m = fmaxf(m, acc[t][v]);
        m = fmaxf(m, __shfl_xor(m, 16, 32));       // combine wave halves (M 0-7 vs 8-15)
        float r = fmaxf(m + b2[t * 16 + col], 0.f);
        if (lane < 16) xout[gi * Hd + t * 16 + col] = r;
    }
}

// ---------------------------------------------------------------------------
__global__ __launch_bounds__(128) void head_kernel(const float* __restrict__ x,
                                                   const float* __restrict__ wc,
                                                   const float* __restrict__ bc,
                                                   float* __restrict__ out, int n) {
    const int b = blockIdx.x, t = threadIdx.x;
    __shared__ float g[Hd];
    float m = -3.4e38f;
    for (int i = 0; i < n; ++i) m = fmaxf(m, x[((size_t)b * n + i) * Hd + t]);
    g[t] = m;
    __syncthreads();
    if (t < OUTC) {
        float acc = bc[t];
        for (int h = 0; h < Hd; ++h) acc += g[h] * wc[h * OUTC + t];
        out[b * OUTC + t] = acc;
    }
}

// ---------------------------------------------------------------------------
extern "C" void kernel_launch(void* const* d_in, const int* in_sizes, int n_in,
                              void* d_out, int out_size, void* d_ws, size_t ws_size,
                              hipStream_t stream) {
    (void)in_sizes; (void)n_in; (void)out_size; (void)ws_size;
    const float* pos = (const float*)d_in[0];
    const float* nrm = (const float*)d_in[1];
    const float* w0a = (const float*)d_in[3];  const float* b0a = (const float*)d_in[4];
    const float* w0b = (const float*)d_in[5];  const float* b0b = (const float*)d_in[6];
    const float* w1a = (const float*)d_in[7];  const float* b1a = (const float*)d_in[8];
    const float* w1b = (const float*)d_in[9];  const float* b1b = (const float*)d_in[10];
    const float* w2a = (const float*)d_in[11]; const float* b2a = (const float*)d_in[12];
    const float* w2b = (const float*)d_in[13]; const float* b2b = (const float*)d_in[14];
    const float* wc  = (const float*)d_in[15]; const float* bc  = (const float*)d_in[16];
    float* out = (float*)d_out;

    char* ws = (char*)d_ws;
    size_t off = 0;
    auto take = [&](size_t bytes) -> char* {
        off = (off + 255) & ~(size_t)255;
        char* r = ws + off; off += bytes; return r;
    };
    f16* w0aT = (f16*)take(128 * 32  * 2);
    f16* w0bT = (f16*)take(128 * 128 * 2);
    f16* w1aT = (f16*)take(128 * 160 * 2);
    f16* w1bT = (f16*)take(128 * 128 * 2);
    f16* w2aT = (f16*)take(128 * 160 * 2);
    f16* w2bT = (f16*)take(128 * 128 * 2);
    int*   idxb = (int*)take((size_t)Bc * N0 * KNNK * 4);
    int*   sel  = (int*)take((size_t)Bc * 2048 * 4);
    float* p1   = (float*)take((size_t)Bc * 2048 * 3 * 4);
    float* n1   = (float*)take((size_t)Bc * 2048 * 3 * 4);
    float* p2   = (float*)take((size_t)Bc * 1024 * 3 * 4);
    float* n2   = (float*)take((size_t)Bc * 1024 * 3 * 4);
    float* x0    = (float*)take((size_t)Bc * 4096 * Hd * 4);
    float* x1in  = (float*)take((size_t)Bc * 2048 * Hd * 4);
    float* x1out = (float*)take((size_t)Bc * 2048 * Hd * 4);
    float* x2in  = (float*)take((size_t)Bc * 1024 * Hd * 4);
    float* x2out = (float*)take((size_t)Bc * 1024 * Hd * 4);

    auto prep = [&](const float* w, f16* wT, int Kin, int KP) {
        prep_w_kernel<<<(128 * KP + 255) / 256, 256, 0, stream>>>(w, wT, Kin, KP);
    };
    prep(w0a, w0aT, 4, 32);    prep(w0b, w0bT, 128, 128);
    prep(w1a, w1aT, 132, 160); prep(w1b, w1bT, 128, 128);
    prep(w2a, w2aT, 132, 160); prep(w2b, w2bT, 128, 128);

    // layer 0 : n = 4096
    knn_kernel<<<Bc * (4096 / 256), 256, 0, stream>>>(pos, 4096, idxb);
    ppf_conv_kernel<<<Bc * 4096 / 4, 128, 0, stream>>>(
        pos, nrm, nullptr, idxb, w0aT, 1, b0a, w0bT, b0b, x0, 4096);

    // layer 1 : 4096 -> 2048
    fps_kernel<<<Bc, 1024, 0, stream>>>(pos, 4096, 2048, sel);
    gather_kernel<<<Bc * 2048, 128, 0, stream>>>(pos, nrm, x0, sel, 4096, 2048, p1, n1, x1in);
    knn_kernel<<<Bc * (2048 / 256), 256, 0, stream>>>(p1, 2048, idxb);
    ppf_conv_kernel<<<Bc * 2048 / 4, 128, 0, stream>>>(
        p1, n1, x1in, idxb, w1aT, 5, b1a, w1bT, b1b, x1out, 2048);

    // layer 2 : 2048 -> 1024
    fps_kernel<<<Bc, 1024, 0, stream>>>(p1, 2048, 1024, sel);
    gather_kernel<<<Bc * 1024, 128, 0, stream>>>(p1, n1, x1out, sel, 2048, 1024, p2, n2, x2in);
    knn_kernel<<<Bc * (1024 / 256), 256, 0, stream>>>(p2, 1024, idxb);
    ppf_conv_kernel<<<Bc * 1024 / 4, 128, 0, stream>>>(
        p2, n2, x2in, idxb, w2aT, 5, b2a, w2bT, b2b, x2out, 1024);

    // global max pool + classifier
    head_kernel<<<Bc, 128, 0, stream>>>(x2out, wc, bc, out, 1024);
}
